// Encoder_39883066311110
// MI455X (gfx1250) — matrix operand order repaired and checked by reading, compiled, not run on hardware
//
#include <hip/hip_runtime.h>
#include <hip/hip_bf16.h>
#include <math.h>

// ---------------------------------------------------------------------------
// MI455X (gfx1250) transformer encoder. All matmuls via v_wmma_f32_16x16x32_f16
// (f16 operands, fp32 accumulate), flash attention (S x S scores never in HBM).
// f16 end-to-end dataflow: weights pre-converted once, activations produced in
// f16, V pre-transposed per layer, so ALL inner-loop staging is pure
// b128 load -> b128 store. fp32 kept for accumulators, softmax, LN stats,
// residual stream. LDS tiles live in *fragment order*: every WMMA operand is
// one contiguous 32-byte v16h load (2x ds_load_b128).
// ---------------------------------------------------------------------------

typedef __attribute__((ext_vector_type(16))) _Float16 v16h;
typedef __attribute__((ext_vector_type(8)))  _Float16 v8h;
typedef __attribute__((ext_vector_type(4)))  _Float16 v4h;
typedef __attribute__((ext_vector_type(8)))  float    v8f;

union H16 { v16h v; _Float16 h[16]; };
union H8  { v8h  v; _Float16 h[8]; };
union H4  { v4h  v; _Float16 h[4]; };

#define D_MODEL 512
#define SEQ     2048
#define NH      8        // num_head (fixed by problem size -> DH = 64)
#define DH      64

#define WMMA_F16(A, B, C) \
  __builtin_amdgcn_wmma_f32_16x16x32_f16(false, (A), false, (B), (short)0, (C), false, false)

// ---------------------------------------------------------------------------
// fp32 -> f16 bulk convert (weights; once per call)
// ---------------------------------------------------------------------------
__global__ void cvt_f16_kernel(const float* __restrict__ in,
                               _Float16* __restrict__ out, int n4) {
  int i = blockIdx.x * blockDim.x + threadIdx.x;
  if (i >= n4) return;
  float4 f = ((const float4*)in)[i];
  H4 u;
  u.h[0] = (_Float16)f.x; u.h[1] = (_Float16)f.y;
  u.h[2] = (_Float16)f.z; u.h[3] = (_Float16)f.w;
  ((v4h*)out)[i] = u.v;
}

// ---------------------------------------------------------------------------
// Embedding gather * sqrt(D) + sinusoidal positional encoding (fp32 out)
// ---------------------------------------------------------------------------
__global__ void embed_pe_kernel(const int* __restrict__ x,
                                const float* __restrict__ embed,
                                float* __restrict__ h, int total) {
  int i = blockIdx.x * blockDim.x + threadIdx.x;
  if (i >= total) return;
  int row = i / D_MODEL, d = i - row * D_MODEL;
  int s = row % SEQ;
  int tok = x[row];
  float v = embed[(size_t)tok * D_MODEL + d] * 22.627416997969522f; // sqrt(512)
  int p = d >> 1;
  float div = __expf((float)(2 * p) * (-9.210340371976184f / (float)D_MODEL));
  float ang = (float)s * div;
  v += (d & 1) ? __cosf(ang) : __sinf(ang);
  h[i] = v;
}

// ---------------------------------------------------------------------------
// LayerNorm: one wave32 per row of 512, fp32 in, f16 or fp32 out
// ---------------------------------------------------------------------------
template <bool OUTH>
__global__ void ln_kernel(const float* __restrict__ in, void* __restrict__ out,
                          const float* __restrict__ g, const float* __restrict__ b,
                          int nrows) {
  int row  = (int)((blockIdx.x * blockDim.x + threadIdx.x) >> 5);
  int lane = threadIdx.x & 31;
  if (row >= nrows) return;
  const float4* r4 = (const float4*)(in + (size_t)row * D_MODEL) + lane * 4;
  float4 v4[4];
  float s = 0.f;
#pragma unroll
  for (int i = 0; i < 4; i++) {
    v4[i] = r4[i];
    s += v4[i].x + v4[i].y + v4[i].z + v4[i].w;
  }
#pragma unroll
  for (int m = 1; m < 32; m <<= 1) s += __shfl_xor(s, m, 32);
  float mean = s * (1.0f / D_MODEL);
  float q = 0.f;
#pragma unroll
  for (int i = 0; i < 4; i++) {
    float dx = v4[i].x - mean, dy = v4[i].y - mean, dz = v4[i].z - mean, dw = v4[i].w - mean;
    q += dx * dx + dy * dy + dz * dz + dw * dw;
  }
#pragma unroll
  for (int m = 1; m < 32; m <<= 1) q += __shfl_xor(q, m, 32);
  float rstd = rsqrtf(q * (1.0f / D_MODEL) + 1e-5f);
  const float4* g4 = (const float4*)g + lane * 4;
  const float4* b4 = (const float4*)b + lane * 4;
  if (OUTH) {
    H16 o16;
#pragma unroll
    for (int i = 0; i < 4; i++) {
      float4 gg = g4[i], bb = b4[i];
      o16.h[i * 4 + 0] = (_Float16)((v4[i].x - mean) * rstd * gg.x + bb.x);
      o16.h[i * 4 + 1] = (_Float16)((v4[i].y - mean) * rstd * gg.y + bb.y);
      o16.h[i * 4 + 2] = (_Float16)((v4[i].z - mean) * rstd * gg.z + bb.z);
      o16.h[i * 4 + 3] = (_Float16)((v4[i].w - mean) * rstd * gg.w + bb.w);
    }
    *(v16h*)((_Float16*)out + (size_t)row * D_MODEL + lane * 16) = o16.v;
  } else {
    float4* o4 = (float4*)((float*)out + (size_t)row * D_MODEL) + lane * 4;
#pragma unroll
    for (int i = 0; i < 4; i++) {
      float4 gg = g4[i], bb = b4[i], o;
      o.x = (v4[i].x - mean) * rstd * gg.x + bb.x;
      o.y = (v4[i].y - mean) * rstd * gg.y + bb.y;
      o.z = (v4[i].z - mean) * rstd * gg.z + bb.z;
      o.w = (v4[i].w - mean) * rstd * gg.w + bb.w;
      o4[i] = o;
    }
  }
}

// ---------------------------------------------------------------------------
// Per-head transpose V[key][dh] -> VT[dh][key], f16, 64x64 LDS tiles.
// Coalesced vector loads, scattered b16 LDS stores (no stalls), contiguous
// LDS reads, coalesced vector stores. Once per layer.
// ---------------------------------------------------------------------------
__global__ __launch_bounds__(128)
void transpose_v_kernel(const _Float16* __restrict__ V, _Float16* __restrict__ VT) {
  __shared__ _Float16 T[64][72];       // [dh][key], padded
  int tid = threadIdx.x;
  int head = blockIdx.x >> 5;          // SEQ/64 == 32 key-tiles per head
  int kt = (blockIdx.x & 31) * 64;
  const _Float16* Vh = V + (size_t)head * SEQ * DH;
  _Float16* VTh = VT + (size_t)head * SEQ * DH;
  {
    int r = tid >> 1, half = tid & 1;  // key row, dh half (32 elems)
    H16 a, b;
    a.v = *(const v16h*)(Vh + (size_t)(kt + r) * DH + half * 32);
    b.v = *(const v16h*)(Vh + (size_t)(kt + r) * DH + half * 32 + 16);
#pragma unroll
    for (int i = 0; i < 16; i++) T[half * 32 + i][r] = a.h[i];
#pragma unroll
    for (int i = 0; i < 16; i++) T[half * 32 + 16 + i][r] = b.h[i];
  }
  __syncthreads();
  {
    int d = tid >> 1, seg = tid & 1;   // dh row, key half
    v16h o0 = *(const v16h*)&T[d][seg * 32];
    v16h o1 = *(const v16h*)&T[d][seg * 32 + 16];
    *(v16h*)(VTh + (size_t)d * SEQ + kt + seg * 32) = o0;
    *(v16h*)(VTh + (size_t)d * SEQ + kt + seg * 32 + 16) = o1;
  }
}

// ---------------------------------------------------------------------------
// WMMA GEMM: out[M,512] = epi(A[M,512](f16) * W[512,512]^T(f16) + bias (+res))
// Block: 128 threads (4 waves). Tile 128(m) x 64(n), K-step 64.
// Staging is pure b128 load -> b128 store (zero VALU). Each wave:
// 2 m-tiles x 4 n-tiles = 16 WMMAs per k-iter on 12 v16h fragment loads.
// MODE: 0 = none, 1 = LeakyReLU(0.01), 2 = x2 (ffn + ffn). OUTH: f16 output.
// ---------------------------------------------------------------------------
template <int MODE, bool RES, bool OUTH>
__global__ __launch_bounds__(128)
void gemm_wmma_kernel(const _Float16* __restrict__ A, const _Float16* __restrict__ W,
                      const float* __restrict__ bias, const _Float16* __restrict__ res,
                      void* __restrict__ out) {
  __shared__ v16h AfT[2][8][32];   // [k-chunk][mtile][lane] : exact A fragments
  __shared__ v16h BfT[2][4][32];   // [k-chunk][ntile][lane] : exact B fragments
  int tid = threadIdx.x;
  int wid = tid >> 5, lane = tid & 31;
  int hi = lane >> 4, lo = lane & 15;
  int bm = blockIdx.x * 128;
  int bn = blockIdx.y * 64;
  int an = tid >> 1, ak = tid & 1;     // B staging: col, k-half
  v8f c[2][4] = {};

  for (int k0 = 0; k0 < 512; k0 += 64) {
    __syncthreads();
    {
      // A: thread = row (tid), 64 consecutive f16 k: 8x (b128 load -> b128 store)
      const v8h* ap = (const v8h*)(A + (size_t)(bm + tid) * 512 + k0);
#pragma unroll
      for (int seg = 0; seg < 8; seg++) {       // k = seg*8 .. +7
        v8h u = ap[seg];
        int cc = seg >> 2, g = seg & 1, hh = (seg >> 1) & 1;
        ((v8h*)&AfT[cc][tid >> 4][hh * 16 + (tid & 15)])[g] = u;
      }
      // B: thread = (col an, k-half ak): W[bn+an][k0+ak*32 .. +31] contiguous f16.
      const v16h* wp = (const v16h*)(W + (size_t)(bn + an) * 512 + k0 + ak * 32);
      BfT[ak][an >> 4][(an & 15)]      = wp[0];
      BfT[ak][an >> 4][16 + (an & 15)] = wp[1];
    }
    __syncthreads();

#pragma unroll
    for (int cc = 0; cc < 2; cc++) {
      v16h a0 = AfT[cc][wid * 2 + 0][lane];
      v16h a1 = AfT[cc][wid * 2 + 1][lane];
      v16h b0 = BfT[cc][0][lane];
      v16h b1 = BfT[cc][1][lane];
      v16h b2 = BfT[cc][2][lane];
      v16h b3 = BfT[cc][3][lane];
      c[0][0] = WMMA_F16(a0, b0, c[0][0]);
      c[0][1] = WMMA_F16(a0, b1, c[0][1]);
      c[0][2] = WMMA_F16(a0, b2, c[0][2]);
      c[0][3] = WMMA_F16(a0, b3, c[0][3]);
      c[1][0] = WMMA_F16(a1, b0, c[1][0]);
      c[1][1] = WMMA_F16(a1, b1, c[1][1]);
      c[1][2] = WMMA_F16(a1, b2, c[1][2]);
      c[1][3] = WMMA_F16(a1, b3, c[1][3]);
    }
  }

  // C layout: element j -> row = hi*8 + j, col = lo. Straight-line epilogue.
#pragma unroll
  for (int mt = 0; mt < 2; mt++) {
    int m0 = bm + (wid * 2 + mt) * 16 + hi * 8;
#pragma unroll
    for (int t = 0; t < 4; t++) {
      int n = bn + t * 16 + lo;
      float bv = bias[n];
      const _Float16* rp = res + (size_t)m0 * 512 + n;
#pragma unroll
      for (int j = 0; j < 8; j++) {
        float v = c[mt][t][j] + bv;
        if (RES) v += (float)rp[j * 512];
        if (MODE == 1) v = (v >= 0.f) ? v : 0.01f * v;
        if (MODE == 2) v *= 2.f;
        if (OUTH) ((_Float16*)out)[(size_t)(m0 + j) * 512 + n] = (_Float16)v;
        else      ((float*)out)[(size_t)(m0 + j) * 512 + n] = v;
      }
    }
  }
}

// ---------------------------------------------------------------------------
// Flash attention, transposed compute, all-f16. Head (b,h) = contiguous
// [SEQ,DH] chunk (bug-faithful reshape); V supplied pre-transposed [DH][SEQ].
// Per wave: 16 queries (q = q0 + lo), full dh=64.
//   S^T = K * Q^T   (A = K tile 16x32-dh, B = Q^T)  -> C col = query = lo
//   O^T = V^T * P^T (A = V^T 16dh x 32key, B = P^T) -> C col = query = lo
// Softmax stats (m_q, l_q) are per-lane scalars; ONE shfl_xor(16) per 32-key
// block. Row sums via ones-WMMA (column sums of P^T). All staging is pure
// vector load -> vector store.
// ---------------------------------------------------------------------------
__global__ __launch_bounds__(128)
void flash_attn_kernel(const _Float16* __restrict__ Q, const _Float16* __restrict__ K,
                       const _Float16* __restrict__ VT, _Float16* __restrict__ O) {
  __shared__ v16h Qf[4][2][32];   // [wid][dh-chunk][lane] : Q^T B-fragments
  __shared__ v16h Kf[2][2][32];   // [keytile][dh-chunk][lane] : K A-fragments
  __shared__ v16h Vf[4][32];      // [dh-tile][lane] : V^T A-fragments
  __shared__ v16h Pf[4][32];      // [wid][lane] : P^T B-fragments
  int tid = threadIdx.x, wid = tid >> 5, lane = tid & 31;
  int hi = lane >> 4, lo = lane & 15;

  int blk = blockIdx.x;           // grid = (B*NH) * (SEQ/64)
  int bh  = blk >> 5;             // SEQ/64 == 32
  size_t base = (size_t)bh * SEQ * DH;
  const _Float16* Qh = Q + base;
  const _Float16* Kh = K + base;
  const _Float16* VTh = VT + base;   // [DH][SEQ]
  _Float16* Oh = O + base;
  int qblk = (blk & 31) * 64;
  int q0 = qblk + wid * 16;

  // all-ones A matrix for column-sum WMMA
  H16 onesU;
#pragma unroll
  for (int i = 0; i < 16; i++) onesU.h[i] = (_Float16)1.0f;
  const v16h onesA = onesU.v;

  // ---- stage Q^T B-fragments once: Qf[wid][c][lane] = Q[q0+lo][c*32+hi*16..]
  {
    int qrow = tid >> 1, sub = tid & 1;
    const v16h* qp = (const v16h*)(Qh + (size_t)(qblk + qrow) * DH + sub * 32);
    Qf[qrow >> 4][sub][(qrow & 15)]      = qp[0];
    Qf[qrow >> 4][sub][16 + (qrow & 15)] = qp[1];
  }
  __syncthreads();
  v16h qb0 = Qf[wid][0][lane];
  v16h qb1 = Qf[wid][1][lane];

  v8f acc[4] = {};                 // O^T: [dh-tile]; elem j -> dh hi*8+j, col lo
  float m_q = -1e30f, l_q = 0.f;   // per-lane stats for query q0+lo
  const float scale = 1.0f / 32.0f; // faithful: dh**1/2 == dh/2 == 32

  for (int kb = 0; kb < SEQ; kb += 32) {
    __syncthreads();
    {
      // K A-fragments: thread -> key rk, 16-dh segment seg: 2x(b128 ld->st).
      int rk = tid >> 2, seg = tid & 3;
      int chunk = seg >> 1, g = seg & 1;
      const v8h* kp = (const v8h*)(Kh + (size_t)(kb + rk) * DH + seg * 16);
      ((v8h*)&Kf[rk >> 4][chunk][g * 16 + (rk & 15)])[0] = kp[0];
      ((v8h*)&Kf[rk >> 4][chunk][g * 16 + (rk & 15)])[1] = kp[1];
      // V^T A-fragments from pre-transposed VT: 2x (b128 load -> b128 store).
      // Vf[t][vg*16+lo].h[g*8+w] = VT[dcol][kb + vg*16 + g*8 + w]
      int dcol = tid >> 1, vg = tid & 1;
      const _Float16* vrow = VTh + (size_t)dcol * SEQ + kb + vg * 16;
      v8h u0 = *(const v8h*)(vrow);
      v8h u1 = *(const v8h*)(vrow + 8);
      ((v8h*)&Vf[dcol >> 4][vg * 16 + (dcol & 15)])[0] = u0;
      ((v8h*)&Vf[dcol >> 4][vg * 16 + (dcol & 15)])[1] = u1;
    }
    __syncthreads();

    // S^T: sT0 = keys kb..kb+15, sT1 = keys kb+16..kb+31 (col = query lo)
    v8f sT0 = {}, sT1 = {};
    {
      v16h ka00 = Kf[0][0][lane], ka01 = Kf[0][1][lane];
      v16h ka10 = Kf[1][0][lane], ka11 = Kf[1][1][lane];
      sT0 = WMMA_F16(ka00, qb0, sT0);
      sT0 = WMMA_F16(ka01, qb1, sT0);
      sT1 = WMMA_F16(ka10, qb0, sT1);
      sT1 = WMMA_F16(ka11, qb1, sT1);
    }

    // softmax for query lo: this lane holds keys {hi*8+j} u {16+hi*8+j}.
    float t0[8], t1[8];
#pragma unroll
    for (int j = 0; j < 8; j++) { t0[j] = sT0[j] * scale; t1[j] = sT1[j] * scale; }
    float mx = fmaxf(t0[0], t1[0]);
#pragma unroll
    for (int j = 1; j < 8; j++) mx = fmaxf(mx, fmaxf(t0[j], t1[j]));
    mx = fmaxf(mx, __shfl_xor(mx, 16, 32));   // combine the two key halves
    float mn = fmaxf(m_q, mx);
    float cr = __expf(m_q - mn);
    m_q = mn;
    float p0[8], p1[8];
#pragma unroll
    for (int j = 0; j < 8; j++) {
      p0[j] = __expf(t0[j] - mn);
      p1[j] = __expf(t1[j] - mn);
      acc[0][j] *= cr; acc[1][j] *= cr; acc[2][j] *= cr; acc[3][j] *= cr;
    }

    // P^T B-fragment scatter: two CONTIGUOUS v8h stores per lane.
    // value(key k, query lo): elem = k&15, lane' = (k>>4)*16 + lo
    {
      H8 u0, u1;
#pragma unroll
      for (int j = 0; j < 8; j++) { u0.h[j] = (_Float16)p0[j]; u1.h[j] = (_Float16)p1[j]; }
      v8h* ph8 = (v8h*)&Pf[wid][0];
      ph8[lo * 2 + hi]      = u0.v;   // keys hi*8..hi*8+7       -> lane' = lo
      ph8[32 + lo * 2 + hi] = u1.v;   // keys 16+hi*8..16+hi*8+7 -> lane' = 16+lo
    }
    __syncthreads();

    v16h pb = Pf[wid][lane];
    // column sums of P^T (= per-query sums) via ones-WMMA: rs[j] all equal.
    v8f rs = {};
    rs = WMMA_F16(onesA, pb, rs);
    v16h va0 = Vf[0][lane], va1 = Vf[1][lane], va2 = Vf[2][lane], va3 = Vf[3][lane];
    acc[0] = WMMA_F16(va0, pb, acc[0]);
    acc[1] = WMMA_F16(va1, pb, acc[1]);
    acc[2] = WMMA_F16(va2, pb, acc[2]);
    acc[3] = WMMA_F16(va3, pb, acc[3]);
    l_q = l_q * cr + rs[0];
  }

  // O^T -> O (f16): lane writes query row q0+lo, dh = t*16 + hi*8 + (0..7)
  {
    float inv = 1.0f / l_q;
    _Float16* orow = Oh + (size_t)(q0 + lo) * DH;
#pragma unroll
    for (int t = 0; t < 4; t++) {
      H8 u;
#pragma unroll
      for (int j = 0; j < 8; j++) u.h[j] = (_Float16)(acc[t][j] * inv);
      *(v8h*)(orow + t * 16 + hi * 8) = u.v;
    }
  }
}

// ---------------------------------------------------------------------------
extern "C" void kernel_launch(void* const* d_in, const int* in_sizes, int n_in,
                              void* d_out, int out_size, void* d_ws, size_t ws_size,
                              hipStream_t stream) {
  (void)n_in; (void)out_size; (void)ws_size;
  const int*   x     = (const int*)d_in[0];
  // d_in[1] = num_head (device scalar; == 8 for this problem -> DH = 64)
  const float* embed = (const float*)d_in[2];
  const float* ln1_g = (const float*)d_in[3];
  const float* ln1_b = (const float*)d_in[4];
  const float* Wq = (const float*)d_in[5];   const float* bq = (const float*)d_in[6];
  const float* Wk = (const float*)d_in[7];   const float* bk = (const float*)d_in[8];
  const float* Wv = (const float*)d_in[9];   const float* bv = (const float*)d_in[10];
  const float* Wo = (const float*)d_in[11];  const float* bo = (const float*)d_in[12];
  const float* ln2_g = (const float*)d_in[13];
  const float* ln2_b = (const float*)d_in[14];
  const float* W1 = (const float*)d_in[15];  const float* b1 = (const float*)d_in[16];
  const float* W2 = (const float*)d_in[17];  const float* b2 = (const float*)d_in[18];
  const float* lnf_g = (const float*)d_in[19];
  const float* lnf_b = (const float*)d_in[20];

  const int BS = in_sizes[0];                 // B*S tokens (8192)
  const int L  = in_sizes[3] / D_MODEL;       // layers from ln1_g = [L,D]
  const size_t nElem = (size_t)BS * D_MODEL;
  const size_t BUF = nElem * sizeof(float);   // one activation buffer (bytes)

  char* wsb = (char*)d_ws;
  float*     hbuf = (float*)(wsb + 0 * BUF);      // fp32 residual stream
  _Float16*  xnh  = (_Float16*)(wsb + 1 * BUF);   // LN outputs (f16)
  _Float16*  qbh  = (_Float16*)(wsb + 2 * BUF);
  _Float16*  kbh  = (_Float16*)(wsb + 3 * BUF);
  _Float16*  vbh  = (_Float16*)(wsb + 4 * BUF);
  _Float16*  abh  = (_Float16*)(wsb + 5 * BUF);   // attention context (f16)
  float*     xr32 = (float*)(wsb + 6 * BUF);      // post-attn residual (fp32)
  _Float16*  vth  = (_Float16*)(wsb + 7 * BUF);   // V transposed per head (f16)
  _Float16*  wgt  = (_Float16*)(wsb + 8 * BUF);   // f16 weights (6 * L*D*D)
  const size_t WSZ = (size_t)L * D_MODEL * D_MODEL;
  _Float16 *Wqh = wgt, *Wkh = wgt + WSZ, *Wvh = wgt + 2 * WSZ,
           *Woh = wgt + 3 * WSZ, *W1h = wgt + 4 * WSZ, *W2h = wgt + 5 * WSZ;

  // one-time (per call) weight conversion fp32 -> f16
  {
    int n4 = (int)(WSZ / 4), blk = (n4 + 255) / 256;
    cvt_f16_kernel<<<blk, 256, 0, stream>>>(Wq, Wqh, n4);
    cvt_f16_kernel<<<blk, 256, 0, stream>>>(Wk, Wkh, n4);
    cvt_f16_kernel<<<blk, 256, 0, stream>>>(Wv, Wvh, n4);
    cvt_f16_kernel<<<blk, 256, 0, stream>>>(Wo, Woh, n4);
    cvt_f16_kernel<<<blk, 256, 0, stream>>>(W1, W1h, n4);
    cvt_f16_kernel<<<blk, 256, 0, stream>>>(W2, W2h, n4);
  }

  {
    int total = (int)nElem;
    embed_pe_kernel<<<(total + 255) / 256, 256, 0, stream>>>(x, embed, hbuf, total);
  }

  dim3 gGemm(BS / 128, D_MODEL / 64);
  dim3 gAttn(BS * NH / 64);                   // (B*NH)*(SEQ/64); also transpose grid
  int  gLN = (BS + 7) / 8;                    // 8 rows per 256-thread block

  for (int l = 0; l < L; l++) {
    const size_t wOff = (size_t)l * D_MODEL * D_MODEL;
    const size_t vOff = (size_t)l * D_MODEL;
    // xn = LN1(h) -> f16
    ln_kernel<true><<<gLN, 256, 0, stream>>>(hbuf, xnh, ln1_g + vOff, ln1_b + vOff, BS);
    // q,k,v projections (f16 -> f16)
    gemm_wmma_kernel<0, false, true><<<gGemm, 128, 0, stream>>>(xnh, Wqh + wOff, bq + vOff, nullptr, qbh);
    gemm_wmma_kernel<0, false, true><<<gGemm, 128, 0, stream>>>(xnh, Wkh + wOff, bk + vOff, nullptr, kbh);
    gemm_wmma_kernel<0, false, true><<<gGemm, 128, 0, stream>>>(xnh, Wvh + wOff, bv + vOff, nullptr, vbh);
    // per-head V transpose -> [dh][seq]
    transpose_v_kernel<<<gAttn, 128, 0, stream>>>(vbh, vth);
    // attention (bug-faithful head split is a contiguous reinterpretation)
    flash_attn_kernel<<<gAttn, 128, 0, stream>>>(qbh, kbh, vth, abh);
    // O-projection + residual with xn -> xr (fp32)
    gemm_wmma_kernel<0, true, false><<<gGemm, 128, 0, stream>>>(abh, Woh + wOff, bo + vOff, xnh, xr32);
    // LN2(xr) -> f16 (xn buffer is free now)
    ln_kernel<true><<<gLN, 256, 0, stream>>>(xr32, xnh, ln2_g + vOff, ln2_b + vOff, BS);
    // ffn1 + LeakyReLU -> f16 (q buffer is free now)
    gemm_wmma_kernel<1, false, true><<<gGemm, 128, 0, stream>>>(xnh, W1h + wOff, b1 + vOff, nullptr, qbh);
    // ffn2, doubled (f + f) -> next h (fp32)
    gemm_wmma_kernel<2, false, false><<<gGemm, 128, 0, stream>>>(qbh, W2h + wOff, b2 + vOff, nullptr, hbuf);
  }

  // final LayerNorm -> d_out (fp32)
  ln_kernel<false><<<gLN, 256, 0, stream>>>(hbuf, d_out, lnf_g, lnf_b, BS);
}
